// CausalTransformer_67619965108246
// MI455X (gfx1250) — compile-verified, matrix operand order checked by reading, not run on hardware
//
#include <hip/hip_runtime.h>
#include <math.h>

#define DEV __device__ __forceinline__

typedef __attribute__((ext_vector_type(16))) _Float16 v16h;
typedef __attribute__((ext_vector_type(8)))  float    v8f;

union Frag { unsigned int u[8]; v16h v; };

// ---------------- problem constants ----------------
constexpr int CB   = 64;     // batch
constexpr int CN   = 55;     // nodes
constexpr int CD   = 256;    // model dim
constexpr int CH   = 8;      // heads
constexpr int CL   = 12;     // layers
constexpr int CE   = 8;      // experts
constexpr int CNL  = 4;      // expert layers
constexpr int CHID = 1024;   // expert hidden
constexpr int CDFF = 2048;   // ffn hidden
constexpr int CSEQ = 2 * CN;        // 110
constexpr int MTOK = CB * CSEQ;     // 7040
constexpr int MBN  = CB * CN;       // 3520
constexpr int CHD  = CD / CH;       // 32

// ---------------- CDNA5 async-to-LDS staging ----------------
#if defined(__has_builtin)
#  if __has_builtin(__builtin_amdgcn_global_load_async_to_lds_b128)
#    define HAVE_ASYNC_LDS 1
#  endif
#endif
#ifndef HAVE_ASYNC_LDS
#  define HAVE_ASYNC_LDS 0
#endif

typedef int v4i_vs __attribute__((vector_size(16)));   // matches builtin param pointee
#define ASG __attribute__((address_space(1)))
#define ASL __attribute__((address_space(3)))

#if HAVE_ASYNC_LDS
DEV void cp16_async(void* l, const void* g) {
  __builtin_amdgcn_global_load_async_to_lds_b128(
      (ASG v4i_vs*)g,
      (ASL v4i_vs*)l, 0, 0);
}
#else
DEV void cp16_async(void* l, const void* g) { *(uint4*)l = *(const uint4*)g; }
#endif

#if defined(__has_builtin)
#  if __has_builtin(__builtin_amdgcn_s_wait_asynccnt)
#    define ASYNC_WAIT() __builtin_amdgcn_s_wait_asynccnt(0)
#  endif
#endif
#ifndef ASYNC_WAIT
#  define ASYNC_WAIT() asm volatile("s_wait_asynccnt 0" ::: "memory")
#endif

// ---------------- device helpers ----------------
DEV unsigned short f2h(float f) {
  _Float16 h = (_Float16)f;
  return __builtin_bit_cast(unsigned short, h);
}
DEV float gelu_exact(float x) { return 0.5f * x * (1.0f + erff(x * 0.70710678118f)); }
DEV float silu(float x) { return x / (1.0f + __expf(-x)); }
DEV float u01(unsigned int x) {
  x ^= 42u * 0x9E3779B9u;
  x ^= x >> 16; x *= 0x7feb352du;
  x ^= x >> 15; x *= 0x846ca68bu;
  x ^= x >> 16;
  return ((x >> 8) + 1u) * (1.0f / 16777218.0f);
}

// ======================================================================
// weight prep: W f32 [K,N]  ->  WT f16 [N,K]  (batched over blockIdx.z)
// ======================================================================
__global__ __launch_bounds__(256) void k_transpose_h(
    const float* __restrict__ W, unsigned short* __restrict__ WT, int K, int N)
{
  __shared__ unsigned short tile[32][33];
  const float* Wz = W + (size_t)blockIdx.z * K * N;
  unsigned short* Tz = WT + (size_t)blockIdx.z * K * N;
  const int k0 = blockIdx.y * 32, n0 = blockIdx.x * 32;
  const int tx = threadIdx.x & 31, ty = threadIdx.x >> 5;   // 8 row-groups
  for (int kk = ty; kk < 32; kk += 8)
    tile[kk][tx] = f2h(Wz[(size_t)(k0 + kk) * N + n0 + tx]);
  __syncthreads();
  for (int nn = ty; nn < 32; nn += 8)
    Tz[(size_t)(n0 + nn) * K + k0 + tx] = tile[tx][nn];
}

// ======================================================================
// WMMA GEMM: out = act(A_f16[M,K] @ WT_f16[N,K]^T + bias (+ res))
// block tile 64(M) x 128(N), K-step 32; 8 waves, each wave owns 32x32
// double-buffered LDS tiles staged with async-to-LDS B128 copies
// ======================================================================
template<int ACT, bool HAS_RES, bool OUTF, bool OUTH>
__global__ __launch_bounds__(256) void k_wmma_gemm(
    const unsigned short* __restrict__ A,    // f16 [M,K]
    const unsigned short* __restrict__ WT,   // f16 [N,K]
    const float* __restrict__ bias,
    const float* __restrict__ res,
    float* __restrict__ outF,
    unsigned short* __restrict__ outH,
    int M, int Nc, int K)
{
  // padded to 80B rows: async B128 dsts stay 16B aligned, frag reads conflict-free
  __shared__ unsigned int AsU[2][64][20];
  __shared__ unsigned int BsU[2][128][20];

  const int tid  = threadIdx.x;
  const int lane = tid & 31;
  const int wave = tid >> 5;
  const int wm   = (wave >> 2) * 32;      // 0,32
  const int wn   = (wave & 3) * 32;       // 0,32,64,96
  const int half = lane >> 4;
  const int l16  = lane & 15;
  const int m0   = blockIdx.y * 64;
  const int n0   = blockIdx.x * 128;

  // loader coords
  const int arow = tid >> 2;              // 0..63
  const int achk = tid & 3;               // 16B chunk within 64B K-row

  auto stage = [&](int kt, int buf) {
    // A tile: 64 rows x 32 f16 = 4KB -> one B128 per thread
    cp16_async((char*)&AsU[buf][arow][0] + achk * 16,
               A + (size_t)(m0 + arow) * K + kt + achk * 8);
    // B tile: 128 rows x 32 f16 = 8KB -> two B128 per thread
#pragma unroll
    for (int j = 0; j < 2; ++j) {
      const int i = tid + j * 256;
      cp16_async((char*)&BsU[buf][i >> 2][0] + (i & 3) * 16,
                 WT + (size_t)(n0 + (i >> 2)) * K + kt + (i & 3) * 8);
    }
  };

  v8f acc00 = {}, acc01 = {}, acc10 = {}, acc11 = {};

  const int nk = K >> 5;
  stage(0, 0);
  ASYNC_WAIT();
  __syncthreads();

  for (int t = 0; t < nk; ++t) {
    const int cur = t & 1;
    if (t + 1 < nk) stage((t + 1) << 5, (t + 1) & 1);

    Frag a0, a1, b0, b1;
#pragma unroll
    for (int v = 0; v < 8; ++v) {
      const int idx = half * 4 + (v & 3) + ((v >> 2) << 3);
      a0.u[v] = AsU[cur][wm + l16][idx];
      a1.u[v] = AsU[cur][wm + 16 + l16][idx];
      b0.u[v] = BsU[cur][wn + l16][idx];
      b1.u[v] = BsU[cur][wn + 16 + l16][idx];
    }
    acc00 = __builtin_amdgcn_wmma_f32_16x16x32_f16(false, a0.v, false, b0.v, (short)0, acc00, false, false);
    acc01 = __builtin_amdgcn_wmma_f32_16x16x32_f16(false, a0.v, false, b1.v, (short)0, acc01, false, false);
    acc10 = __builtin_amdgcn_wmma_f32_16x16x32_f16(false, a1.v, false, b0.v, (short)0, acc10, false, false);
    acc11 = __builtin_amdgcn_wmma_f32_16x16x32_f16(false, a1.v, false, b1.v, (short)0, acc11, false, false);

    ASYNC_WAIT();
    __syncthreads();
  }

  auto emit = [&](float v, int gm, int gn) {
    if (bias) v += bias[gn];
    if (HAS_RES) v += res[(size_t)gm * Nc + gn];
    if (ACT == 1) v = gelu_exact(v);
    if (OUTF) outF[(size_t)gm * Nc + gn] = v;
    if (OUTH) outH[(size_t)gm * Nc + gn] = f2h(v);
  };
#pragma unroll
  for (int r = 0; r < 8; ++r) {
    const int gm0 = m0 + wm + r + 8 * half;
    const int gm1 = gm0 + 16;
    const int gn0 = n0 + wn + l16;
    const int gn1 = gn0 + 16;
    emit(acc00[r], gm0, gn0); emit(acc01[r], gm0, gn1);
    emit(acc10[r], gm1, gn0); emit(acc11[r], gm1, gn1);
  }
}

// ======================================================================
// token embedding build
// ======================================================================
__global__ __launch_bounds__(256) void k_build_tokens(
    const float* __restrict__ base, const float* __restrict__ ints,
    const float* __restrict__ tgt,  const float* __restrict__ mask,
    const float* __restrict__ feat_emb, const float* __restrict__ valW,
    const float* __restrict__ valb,
    float* __restrict__ x, unsigned short* __restrict__ xh)
{
  const int row = blockIdx.x;
  const int d   = threadIdx.x;
  const int b   = row / CSEQ;
  const int t   = row % CSEQ;
  const int n   = t >> 1;
  float v;
  if ((t & 1) == 0) {
    v = feat_emb[n * CD + d];
  } else {
    const int bn = b * CN + n;
    v = base[bn] * valW[0 * CD + d] + ints[bn] * valW[1 * CD + d] +
        tgt[bn] * valW[2 * CD + d] + mask[bn] * valW[3 * CD + d] + valb[d];
  }
  x[(size_t)row * CD + d]  = v;
  xh[(size_t)row * CD + d] = f2h(v);
}

// ======================================================================
// softmax attention, one (b,h) per block; 2-pass softmax
// ======================================================================
__global__ __launch_bounds__(128) void k_attention(
    const float* __restrict__ qkv, float* __restrict__ o,
    unsigned short* __restrict__ oh)
{
  __shared__ float qs[CSEQ][CHD + 1];
  __shared__ float ks[CSEQ][CHD + 1];
  __shared__ float vs[CSEQ][CHD + 1];
  const int bh = blockIdx.x;
  const int b  = bh / CH;
  const int h  = bh % CH;
  for (int i = threadIdx.x; i < CSEQ * CHD; i += 128) {
    const int s = i >> 5, d = i & 31;
    const size_t rb = (size_t)(b * CSEQ + s) * (3 * CD);
    qs[s][d] = qkv[rb + h * CHD + d];
    ks[s][d] = qkv[rb + CD + h * CHD + d];
    vs[s][d] = qkv[rb + 2 * CD + h * CHD + d];
  }
  __syncthreads();
  const int qi = threadIdx.x;
  if (qi < CSEQ) {
    const float scale = 0.17677669529663687f;  // 1/sqrt(32)
    float mx = -1e30f;
    for (int s = 0; s < CSEQ; ++s) {
      float dot = 0.f;
#pragma unroll
      for (int d = 0; d < CHD; ++d) dot += qs[qi][d] * ks[s][d];
      mx = fmaxf(mx, dot * scale);
    }
    float acc[CHD];
#pragma unroll
    for (int d = 0; d < CHD; ++d) acc[d] = 0.f;
    float se = 0.f;
    for (int s = 0; s < CSEQ; ++s) {
      float dot = 0.f;
#pragma unroll
      for (int d = 0; d < CHD; ++d) dot += qs[qi][d] * ks[s][d];
      const float p = __expf(dot * scale - mx);
      se += p;
#pragma unroll
      for (int d = 0; d < CHD; ++d) acc[d] += p * vs[s][d];
    }
    const float inv = 1.0f / se;
    const size_t ro = (size_t)(b * CSEQ + qi) * CD + h * CHD;
#pragma unroll
    for (int d = 0; d < CHD; ++d) {
      const float val = acc[d] * inv;
      o[ro + d]  = val;
      oh[ro + d] = f2h(val);
    }
  }
}

// ======================================================================
// x = LN(x + add)
// ======================================================================
__global__ __launch_bounds__(256) void k_add_ln(
    float* __restrict__ x, const float* __restrict__ add,
    const float* __restrict__ g, const float* __restrict__ bta,
    unsigned short* __restrict__ xh)
{
  __shared__ float red[256];
  const int row = blockIdx.x;
  const int d = threadIdx.x;
  float v = x[(size_t)row * CD + d] + add[(size_t)row * CD + d];
  red[d] = v; __syncthreads();
  for (int off = 128; off > 0; off >>= 1) {
    if (d < off) red[d] += red[d + off];
    __syncthreads();
  }
  const float mu = red[0] * (1.0f / CD);
  __syncthreads();
  const float dv = v - mu;
  red[d] = dv * dv; __syncthreads();
  for (int off = 128; off > 0; off >>= 1) {
    if (d < off) red[d] += red[d + off];
    __syncthreads();
  }
  const float var = red[0] * (1.0f / CD);
  const float y = g[d] * dv * rsqrtf(var + 1e-5f) + bta[d];
  x[(size_t)row * CD + d]  = y;
  xh[(size_t)row * CD + d] = f2h(y);
}

// RMS norm (eps 1e-8), writes f16 only
__global__ __launch_bounds__(256) void k_rms(
    const float* __restrict__ h, const float* __restrict__ scale,
    unsigned short* __restrict__ hn)
{
  __shared__ float red[256];
  const int row = blockIdx.x;
  const int d = threadIdx.x;
  const float v = h[(size_t)row * CD + d];
  red[d] = v * v; __syncthreads();
  for (int off = 128; off > 0; off >>= 1) {
    if (d < off) red[d] += red[d + off];
    __syncthreads();
  }
  const float ms = red[0] * (1.0f / CD);
  hn[(size_t)row * CD + d] = f2h(scale[d] * v * rsqrtf(ms + 1e-8f));
}

__global__ __launch_bounds__(256) void k_silumul(
    const float* __restrict__ g, const float* __restrict__ v,
    unsigned short* __restrict__ out, int n)
{
  const int i = blockIdx.x * 256 + threadIdx.x;
  if (i < n) out[i] = f2h(silu(g[i]) * v[i]);
}

// feature/value token extraction
__global__ __launch_bounds__(256) void k_extract(
    const float* __restrict__ x, float* __restrict__ feat,
    unsigned short* __restrict__ feath, float* __restrict__ val)
{
  const int bn = blockIdx.x;
  const int d = threadIdx.x;
  const int b = bn / CN, n = bn % CN;
  const float f = x[(size_t)(b * CSEQ + 2 * n) * CD + d];
  feat[(size_t)bn * CD + d]  = f;
  feath[(size_t)bn * CD + d] = f2h(f);
  val[(size_t)bn * CD + d]   = x[(size_t)(b * CSEQ + 2 * n + 1) * CD + d];
}

// hypernetwork: mod[b] = gelu(instr @ W1 + b1) @ W2 + b2
__global__ __launch_bounds__(64) void k_hyper(
    const int* __restrict__ idx, const float* __restrict__ emb,
    const float* __restrict__ W1, const float* __restrict__ b1,
    const float* __restrict__ W2, const float* __restrict__ b2,
    float* __restrict__ mod)
{
  __shared__ float ins[32];
  __shared__ float hid[64];
  const int b = blockIdx.x;
  const int t = threadIdx.x;
  if (t < 32) ins[t] = emb[idx[b] * 32 + t];
  __syncthreads();
  float a = b1[t];
  for (int k = 0; k < 32; ++k) a += ins[k] * W1[k * 64 + t];
  hid[t] = gelu_exact(a);
  __syncthreads();
  for (int d = t; d < CD; d += 64) {
    float s = b2[d];
    for (int k = 0; k < 64; ++k) s += hid[k] * W2[k * CD + d];
    mod[b * CD + d] = s;
  }
}

// x_mod = value_tokens * mod[b]
__global__ __launch_bounds__(256) void k_xmod(
    const float* __restrict__ val, const float* __restrict__ mod,
    float* __restrict__ xm, unsigned short* __restrict__ xmh)
{
  const int bn = blockIdx.x;
  const int d = threadIdx.x;
  const int b = bn / CN;
  const float v = val[(size_t)bn * CD + d] * mod[b * CD + d];
  xm[(size_t)bn * CD + d]  = v;
  xmh[(size_t)bn * CD + d] = f2h(v);
}

// router softmax
__global__ __launch_bounds__(64) void k_router(
    const float* __restrict__ xm, const float* __restrict__ rW,
    const float* __restrict__ rb, float* __restrict__ wgt)
{
  __shared__ float part[64][CE];
  __shared__ float lg[CE];
  const int bn = blockIdx.x;
  const int n = bn % CN;
  const int t = threadIdx.x;
  float acc[CE];
#pragma unroll
  for (int e = 0; e < CE; ++e) acc[e] = 0.f;
  for (int d = t; d < CD; d += 64) {
    const float xv = xm[(size_t)bn * CD + d];
    const float* w = &rW[(size_t)n * CD * CE + (size_t)d * CE];
#pragma unroll
    for (int e = 0; e < CE; ++e) acc[e] += xv * w[e];
  }
#pragma unroll
  for (int e = 0; e < CE; ++e) part[t][e] = acc[e];
  __syncthreads();
  if (t < CE) {
    float s = rb[n * CE + t];
    for (int i = 0; i < 64; ++i) s += part[i][t];
    lg[t] = s;
  }
  __syncthreads();
  if (t == 0) {
    float mx = lg[0];
    for (int e = 1; e < CE; ++e) mx = fmaxf(mx, lg[e]);
    float se = 0.f;
    float ex[CE];
    for (int e = 0; e < CE; ++e) { ex[e] = __expf(lg[e] - mx); se += ex[e]; }
    const float inv = 1.0f / se;
    for (int e = 0; e < CE; ++e) wgt[(size_t)bn * CE + e] = ex[e] * inv;
  }
}

// expert head: out[bn] = h[bn] . fW + fb
__global__ __launch_bounds__(64) void k_expfinal(
    const float* __restrict__ h, const float* __restrict__ fW,
    const float* __restrict__ fb, float* __restrict__ out)
{
  __shared__ float red[64];
  const int bn = blockIdx.x;
  const int t = threadIdx.x;
  float s = 0.f;
  for (int d = t; d < CD; d += 64) s += h[(size_t)bn * CD + d] * fW[d];
  red[t] = s; __syncthreads();
  for (int off = 32; off > 0; off >>= 1) {
    if (t < off) red[t] += red[t + off];
    __syncthreads();
  }
  if (t == 0) out[bn] = red[0] + fb[0];
}

// deltas[bn] = sum_e exp_out[e,bn] * weights[bn,e]
__global__ __launch_bounds__(256) void k_deltas(
    const float* __restrict__ eo, const float* __restrict__ wgt,
    float* __restrict__ dout)
{
  const int bn = blockIdx.x * 256 + threadIdx.x;
  if (bn < MBN) {
    float s = 0.f;
#pragma unroll
    for (int e = 0; e < CE; ++e) s += eo[e * MBN + bn] * wgt[(size_t)bn * CE + e];
    dout[bn] = s;
  }
}

// DAG logits + gumbel-sigmoid adjacency; c block for batch staged in LDS
__global__ __launch_bounds__(64) void k_dag(
    const float* __restrict__ p, const float* __restrict__ c,
    float* __restrict__ logits, float* __restrict__ adj)
{
  __shared__ float ps[CD];
  __shared__ float cs[CN][CD + 1];
  const int bn = blockIdx.x;
  const int b = bn / CN, n = bn % CN;
  for (int i = threadIdx.x; i < CD; i += 64) ps[i] = p[(size_t)bn * CD + i];
  for (int i = threadIdx.x; i < CN * CD; i += 64)
    cs[i >> 8][i & 255] = c[(size_t)b * CN * CD + i];
  __syncthreads();
  const int m = threadIdx.x;
  if (m < CN) {
    float s = 0.f;
    for (int d = 0; d < CD; ++d) s += ps[d] * cs[m][d];
    const int o = (b * CN + n) * CN + m;
    logits[o] = s;
    const float u = u01((unsigned int)o * 0x9E3779B9u + 0x85EBCA6Bu);
    const float gum = -__logf(-__logf(u));
    adj[o] = 1.0f / (1.0f + __expf(-(s + gum)));   // TAU = 1
  }
}

// ======================================================================
// host side
// ======================================================================
static inline void gemm_plain(hipStream_t s, const unsigned short* A, const unsigned short* WT,
                              const float* bias, float* outF, int M, int N, int K) {
  dim3 g(N / 128, M / 64), b(256);
  k_wmma_gemm<0, false, true, false><<<g, b, 0, s>>>(A, WT, bias, nullptr, outF, nullptr, M, N, K);
}
static inline void gemm_gelu_h(hipStream_t s, const unsigned short* A, const unsigned short* WT,
                               const float* bias, unsigned short* outH, int M, int N, int K) {
  dim3 g(N / 128, M / 64), b(256);
  k_wmma_gemm<1, false, false, true><<<g, b, 0, s>>>(A, WT, bias, nullptr, nullptr, outH, M, N, K);
}
static inline void gemm_res(hipStream_t s, const unsigned short* A, const unsigned short* WT,
                            const float* res, float* outF, int M, int N, int K) {
  dim3 g(N / 128, M / 64), b(256);
  k_wmma_gemm<0, true, true, false><<<g, b, 0, s>>>(A, WT, nullptr, res, outF, nullptr, M, N, K);
}
static inline void transpose_h(hipStream_t s, const float* W, unsigned short* WT,
                               int K, int N, int Z) {
  dim3 g(N / 32, K / 32, Z), b(256);
  k_transpose_h<<<g, b, 0, s>>>(W, WT, K, N);
}

extern "C" void kernel_launch(void* const* d_in, const int* in_sizes, int n_in,
                              void* d_out, int out_size, void* d_ws, size_t ws_size,
                              hipStream_t stream) {
  (void)in_sizes; (void)n_in; (void)out_size; (void)ws_size;

  // ---- inputs (setup_inputs dict order, params flattened in dict order) ----
  const float* base    = (const float*)d_in[0];
  const float* ints    = (const float*)d_in[1];
  const float* tgt     = (const float*)d_in[2];
  const float* mask    = (const float*)d_in[3];
  const int*   nodeidx = (const int*)  d_in[4];
  const float* feat_emb= (const float*)d_in[5];
  const float* valW    = (const float*)d_in[6];
  const float* valb    = (const float*)d_in[7];
  const float* Wqkv    = (const float*)d_in[8];
  const float* bqkv    = (const float*)d_in[9];
  const float* Wo      = (const float*)d_in[10];
  const float* bo      = (const float*)d_in[11];
  const float* ln1g    = (const float*)d_in[12];
  const float* ln1b    = (const float*)d_in[13];
  const float* ln2g    = (const float*)d_in[14];
  const float* ln2b    = (const float*)d_in[15];
  const float* W1      = (const float*)d_in[16];
  const float* b1      = (const float*)d_in[17];
  const float* W2      = (const float*)d_in[18];
  const float* b2      = (const float*)d_in[19];
  const float* int_emb = (const float*)d_in[20];
  const float* hypW1   = (const float*)d_in[21];
  const float* hypb1   = (const float*)d_in[22];
  const float* hypW2   = (const float*)d_in[23];
  const float* hypb2   = (const float*)d_in[24];
  const float* routerW = (const float*)d_in[25];
  const float* routerb = (const float*)d_in[26];
  const float* escale  = (const float*)d_in[27];
  const float* ewg     = (const float*)d_in[28];
  const float* ewv     = (const float*)d_in[29];
  const float* ewo     = (const float*)d_in[30];
  const float* efW     = (const float*)d_in[31];
  const float* efb     = (const float*)d_in[32];
  const float* dagpW   = (const float*)d_in[33];
  const float* dagpb   = (const float*)d_in[34];
  const float* dagcW   = (const float*)d_in[35];
  const float* dagcb   = (const float*)d_in[36];

  float* out_deltas = (float*)d_out;
  float* out_logits = out_deltas + MBN;
  float* out_adj    = out_logits + MBN * CN;

  // ---- workspace carve-out ----
  size_t off = 0;
  auto carve = [&](size_t bytes) -> char* {
    char* p = (char*)d_ws + off;
    off += (bytes + 255) & ~(size_t)255;
    return p;
  };
  // activations
  float*          xF    = (float*)         carve((size_t)MTOK * CD * 4);
  unsigned short* xH    = (unsigned short*)carve((size_t)MTOK * CD * 2);
  float*          qkvF  = (float*)         carve((size_t)MTOK * 3 * CD * 4);
  float*          oF    = (float*)         carve((size_t)MTOK * CD * 4);
  unsigned short* oH    = (unsigned short*)carve((size_t)MTOK * CD * 2);
  float*          tmpF  = (float*)         carve((size_t)MTOK * CD * 4);
  unsigned short* h1H   = (unsigned short*)carve((size_t)MTOK * CDFF * 2);
  float*          featF = (float*)         carve((size_t)MBN * CD * 4);
  unsigned short* featH = (unsigned short*)carve((size_t)MBN * CD * 2);
  float*          valF  = (float*)         carve((size_t)MBN * CD * 4);
  float*          modF  = (float*)         carve((size_t)CB * CD * 4);
  float*          xmF   = (float*)         carve((size_t)MBN * CD * 4);
  unsigned short* xmH   = (unsigned short*)carve((size_t)MBN * CD * 2);
  float*          wgt   = (float*)         carve((size_t)MBN * CE * 4);
  float*          hF    = (float*)         carve((size_t)MBN * CD * 4);
  unsigned short* hnH   = (unsigned short*)carve((size_t)MBN * CD * 2);
  float*          gF    = (float*)         carve((size_t)MBN * CHID * 4);
  float*          vF    = (float*)         carve((size_t)MBN * CHID * 4);
  unsigned short* gvH   = (unsigned short*)carve((size_t)MBN * CHID * 2);
  float*          eout  = (float*)         carve((size_t)CE * MBN * 4);
  float*          pF    = (float*)         carve((size_t)MBN * CD * 4);
  float*          cF    = (float*)         carve((size_t)MBN * CD * 4);
  // pre-transposed f16 weights [N,K]
  unsigned short* WqkvT = (unsigned short*)carve((size_t)CL * CD * 3 * CD * 2);
  unsigned short* WoT   = (unsigned short*)carve((size_t)CL * CD * CD * 2);
  unsigned short* W1T   = (unsigned short*)carve((size_t)CL * CD * CDFF * 2);
  unsigned short* W2T   = (unsigned short*)carve((size_t)CL * CDFF * CD * 2);
  unsigned short* ewgT  = (unsigned short*)carve((size_t)CE * CNL * CD * CHID * 2);
  unsigned short* ewvT  = (unsigned short*)carve((size_t)CE * CNL * CD * CHID * 2);
  unsigned short* ewoT  = (unsigned short*)carve((size_t)CE * CNL * CHID * CD * 2);
  unsigned short* dagpT = (unsigned short*)carve((size_t)CD * CD * 2);
  unsigned short* dagcT = (unsigned short*)carve((size_t)CD * CD * 2);

  // ---- 0. one-time weight transpose+convert (runs per call; bandwidth-trivial) ----
  transpose_h(stream, Wqkv, WqkvT, CD, 3 * CD, CL);
  transpose_h(stream, Wo,   WoT,   CD, CD,     CL);
  transpose_h(stream, W1,   W1T,   CD, CDFF,   CL);
  transpose_h(stream, W2,   W2T,   CDFF, CD,   CL);
  transpose_h(stream, ewg,  ewgT,  CD, CHID,   CE * CNL);
  transpose_h(stream, ewv,  ewvT,  CD, CHID,   CE * CNL);
  transpose_h(stream, ewo,  ewoT,  CHID, CD,   CE * CNL);
  transpose_h(stream, dagpW, dagpT, CD, CD, 1);
  transpose_h(stream, dagcW, dagcT, CD, CD, 1);

  // ---- 1. tokens ----
  k_build_tokens<<<MTOK, 256, 0, stream>>>(base, ints, tgt, mask, feat_emb, valW, valb, xF, xH);

  // ---- 2. transformer stack ----
  for (int l = 0; l < CL; ++l) {
    gemm_plain(stream, xH, WqkvT + (size_t)l * CD * 3 * CD, bqkv + (size_t)l * 3 * CD,
               qkvF, MTOK, 3 * CD, CD);
    k_attention<<<CB * CH, 128, 0, stream>>>(qkvF, oF, oH);
    gemm_plain(stream, oH, WoT + (size_t)l * CD * CD, bo + (size_t)l * CD,
               tmpF, MTOK, CD, CD);
    k_add_ln<<<MTOK, 256, 0, stream>>>(xF, tmpF, ln1g + (size_t)l * CD, ln1b + (size_t)l * CD, xH);
    gemm_gelu_h(stream, xH, W1T + (size_t)l * CD * CDFF, b1 + (size_t)l * CDFF,
                h1H, MTOK, CDFF, CD);
    gemm_plain(stream, h1H, W2T + (size_t)l * CDFF * CD, b2 + (size_t)l * CD,
               tmpF, MTOK, CD, CDFF);
    k_add_ln<<<MTOK, 256, 0, stream>>>(xF, tmpF, ln2g + (size_t)l * CD, ln2b + (size_t)l * CD, xH);
  }

  // ---- 3. token split, hypernetwork, modulation, router ----
  k_extract<<<MBN, 256, 0, stream>>>(xF, featF, featH, valF);
  k_hyper<<<CB, 64, 0, stream>>>(nodeidx, int_emb, hypW1, hypb1, hypW2, hypb2, modF);
  k_xmod<<<MBN, 256, 0, stream>>>(valF, modF, xmF, xmH);
  k_router<<<MBN, 64, 0, stream>>>(xmF, routerW, routerb, wgt);

  // ---- 4. experts (h starts as x_mod; first block reads x_mod directly) ----
  for (int e = 0; e < CE; ++e) {
    for (int nl = 0; nl < CNL; ++nl) {
      const size_t li = (size_t)(e * CNL + nl);
      const float* hcur = (nl == 0) ? xmF : hF;
      k_rms<<<MBN, 256, 0, stream>>>(hcur, escale + li * CD, hnH);
      gemm_plain(stream, hnH, ewgT + li * CD * CHID, nullptr, gF, MBN, CHID, CD);
      gemm_plain(stream, hnH, ewvT + li * CD * CHID, nullptr, vF, MBN, CHID, CD);
      k_silumul<<<(MBN * CHID + 255) / 256, 256, 0, stream>>>(gF, vF, gvH, MBN * CHID);
      gemm_res(stream, gvH, ewoT + li * CHID * CD, hcur, hF, MBN, CD, CHID);
    }
    k_expfinal<<<MBN, 64, 0, stream>>>(hF, efW + (size_t)e * CD, efb + e, eout + (size_t)e * MBN);
  }
  k_deltas<<<(MBN + 255) / 256, 256, 0, stream>>>(eout, wgt, out_deltas);

  // ---- 5. DAG head ----
  gemm_plain(stream, featH, dagpT, dagpb, pF, MBN, CD, CD);
  gemm_plain(stream, featH, dagcT, dagcb, cF, MBN, CD, CD);
  k_dag<<<MBN, 64, 0, stream>>>(pF, cF, out_logits, out_adj);
}
